// EdgeEncoder_59742995087902
// MI455X (gfx1250) — matrix operand order repaired and checked by reading, not compiled, not run on hardware
//
#include <hip/hip_runtime.h>
#include <stdint.h>

// ---------------- constants from the reference ----------------
constexpr int kE     = 65536;
constexpr int kD     = 512;
constexpr int kPMAX  = 4;
constexpr int kNBASE = 9;
constexpr int kNTYPE = 14;

constexpr int kROWS  = 64;          // edges per block (4 M-tiles of 16)
constexpr int kHSTR  = 520;         // LDS stride (halves) for h   [64][520]
constexpr int kCSTR  = 1544;        // LDS stride (halves) for concat [64][1544]
constexpr int kTILES_PER_BASE = (kE + kROWS - 1) / kROWS;   // 1024

__constant__ int c_type2base[kNTYPE] = {0,0,0,1,1,1,2,2,3,4,5,6,7,8};
__constant__ float c_scales[kNBASE][kPMAX] = {
  {1.f,    1e-6f, 1.f,  1.f},   // nmos
  {1.f,    1e-6f, 1.f,  1.f},   // pmos
  {1.f,    1.f,   1.f,  1.f},   // balun
  {1000.f, 1.f,   1.f,  1.f},   // resistor
  {1e-12f, 1.f,   1.f,  1.f},   // capacitor
  {1e-9f,  1.f,   1.f,  1.f},   // inductor
  {1.f,    1.f,   1.f,  1.f},   // vsource
  {1e-3f,  1e-3f, 1.f,  1.f},   // isource
  {1.f,    1.f,   1e9f, 1.f},   // port
};

typedef __attribute__((ext_vector_type(16))) __bf16 v16bf;
typedef __attribute__((ext_vector_type(8)))  float  v8f;

union Frag {            // one WMMA A/B bf16 operand: 16 halves / lane
  v16bf v;
  uint4 q[2];
};

__device__ __forceinline__ unsigned short f32_to_bf16(float f) {
  unsigned u = __builtin_bit_cast(unsigned, f);
  u += 0x7FFFu + ((u >> 16) & 1u);          // round to nearest even
  return (unsigned short)(u >> 16);
}

// ---------------- prep kernels ----------------
__global__ void zero_counts_kernel(int* counts) {
  if (threadIdx.x < 16) counts[threadIdx.x] = 0;
}

__global__ void bin_edges_kernel(const int* __restrict__ type_ids,
                                 int* __restrict__ counts,
                                 int* __restrict__ buckets) {
  int i = blockIdx.x * 256 + threadIdx.x;
  if (i >= kE) return;
  int b = c_type2base[type_ids[i]];
  int pos = atomicAdd(&counts[b], 1);
  buckets[b * kE + pos] = i;
}

// W2 [9][512][512] f32 -> bf16, pre-swizzled into WMMA B fragments:
// block id = (t*32 + ntile)*16 + ktile ; within block: lane*16 + h where
// lane = (k>=ktile*32+16)*16 + (n&15), h = k % 16  (k order used here, same as A: lanes 0-15 K 0-15,
// lanes 16-31 K 16-31, column per lane).
__global__ void swz_w2_kernel(const float* __restrict__ W2, unsigned short* __restrict__ out) {
  int idx = blockIdx.x * 256 + threadIdx.x;
  if (idx >= kNBASE * 32 * 16 * 512) return;
  int h16  = idx & 511;
  int blk  = idx >> 9;
  int lane = h16 >> 4, h = h16 & 15;
  int kt   = blk & 15;
  int rest = blk >> 4;
  int nt   = rest & 31;
  int t    = rest >> 5;
  int k = kt * 32 + (lane >> 4) * 16 + h;
  int n = nt * 16 + (lane & 15);
  out[idx] = f32_to_bf16(W2[(t * kD + k) * kD + n]);
}

// Wf [1536][512] f32 -> bf16 fragments: block id = ntile*48 + ktile
__global__ void swz_wf_kernel(const float* __restrict__ Wf, unsigned short* __restrict__ out) {
  int idx = blockIdx.x * 256 + threadIdx.x;
  if (idx >= 32 * 48 * 512) return;
  int h16  = idx & 511;
  int blk  = idx >> 9;
  int lane = h16 >> 4, h = h16 & 15;
  int kt   = blk % 48;
  int nt   = blk / 48;
  int k = kt * 32 + (lane >> 4) * 16 + h;
  int n = nt * 16 + (lane & 15);
  out[idx] = f32_to_bf16(Wf[k * kD + n]);
}

// ---------------- main fused kernel ----------------
__launch_bounds__(256, 1)
__global__ void edge_mlp_kernel(const int*   __restrict__ type_ids,
                                const int*   __restrict__ source_ids,
                                const float* __restrict__ params,
                                const float* __restrict__ type_embed,
                                const float* __restrict__ source_embed,
                                const float* __restrict__ W1,
                                const float* __restrict__ b1,
                                const float* __restrict__ b2,
                                const float* __restrict__ bfv,
                                const int*   __restrict__ counts,
                                const int*   __restrict__ buckets,
                                const unsigned short* __restrict__ W2s,
                                const unsigned short* __restrict__ Wfs,
                                float* __restrict__ out)
{
  extern __shared__ __align__(128) char smem_raw[];
  unsigned short* sC  = (unsigned short*)smem_raw;        // concat bf16 [64][1544]
  unsigned short* sH  = sC + kROWS * kCSTR;               // h bf16      [64][520]
  float*          sX  = (float*)(sH + kROWS * kHSTR);     // x           [64][4]
  int*            sIdx = (int*)(sX + kROWS * kPMAX);      // edge ids    [64]
  int*            sTid = sIdx + kROWS;
  int*            sSid = sTid + kROWS;

  const int base = blockIdx.x >> 10;                      // 1024 tile slots per base
  const int tile = blockIdx.x & (kTILES_PER_BASE - 1);
  const int cnt  = counts[base];
  const int row0 = tile * kROWS;
  if (row0 >= cnt) return;

  const int tid  = threadIdx.x;
  const int lane = tid & 31;
  const int wv   = tid >> 5;                              // 8 waves
  const int n0   = wv * 64;                               // this wave's output columns

  // ---- stage 0: gather edge metadata, scale params ----
  if (tid < kROWS) {
    int r = row0 + tid;
    bool valid = (r < cnt);
    int e = buckets[base * kE + (valid ? r : 0)];
    sIdx[tid] = valid ? e : -1;
    sTid[tid] = type_ids[e];
    sSid[tid] = source_ids[e];
#pragma unroll
    for (int p = 0; p < kPMAX; ++p)
      sX[tid * kPMAX + p] = valid ? params[e * kPMAX + p] / c_scales[base][p] : 0.f;
  }
  __syncthreads();

  // ---- stage 1: h = relu(x @ W1[base] + b1[base])  (K=4, VALU) ----
  for (int i = tid; i < kROWS * kD; i += 256) {
    int e = i >> 9, d = i & (kD - 1);
    float acc = b1[base * kD + d];
#pragma unroll
    for (int p = 0; p < kPMAX; ++p)
      acc += sX[e * kPMAX + p] * W1[(base * kPMAX + p) * kD + d];
    sH[e * kHSTR + d] = f32_to_bf16(fmaxf(acc, 0.f));
  }
  __syncthreads();

  // ---- stage 3: te/se gathers -> concat columns [0,1024) ----
  for (int i = tid; i < kROWS * kD; i += 256) {
    int e = i >> 9, d = i & (kD - 1);
    sC[e * kCSTR + d]      = f32_to_bf16(type_embed[sTid[e] * kD + d]);
    sC[e * kCSTR + kD + d] = f32_to_bf16(source_embed[sSid[e] * kD + d]);
  }

  const int mrow = lane & 15;          // A row within tile
  const int khlf = lane >> 4;          // 0 -> K 0-15, 1 -> K 16-31
  const v8f vzero = {0.f,0.f,0.f,0.f,0.f,0.f,0.f,0.f};

  // ---- stage 2: pv = h @ W2[base] + b2 -> concat columns [1024,1536) ----
  {
    v8f acc[4][4];
#pragma unroll
    for (int mt = 0; mt < 4; ++mt)
#pragma unroll
      for (int nt = 0; nt < 4; ++nt) acc[mt][nt] = vzero;

    for (int kt = 0; kt < kD / 32; ++kt) {
      const int k0 = kt * 32;
      Frag a[4], b[4];
#pragma unroll
      for (int mt = 0; mt < 4; ++mt) {
        const unsigned short* rp = sH + (mt * 16 + mrow) * kHSTR + k0 + khlf * 16;
        a[mt].q[0] = *(const uint4*)(rp);
        a[mt].q[1] = *(const uint4*)(rp + 8);
      }
#pragma unroll
      for (int nt = 0; nt < 4; ++nt) {
        const unsigned short* bp =
            W2s + ((((base * 32) + (wv * 4 + nt)) * 16 + kt) << 9) + lane * 16;
        b[nt].q[0] = *(const uint4*)(bp);
        b[nt].q[1] = *(const uint4*)(bp + 8);
        __builtin_prefetch(bp + 512, 0, 0);   // next K-tile fragment
      }
#pragma unroll
      for (int mt = 0; mt < 4; ++mt)
#pragma unroll
        for (int nt = 0; nt < 4; ++nt)
          acc[mt][nt] = __builtin_amdgcn_wmma_f32_16x16x32_bf16(
              false, a[mt].v, false, b[nt].v, (short)0, acc[mt][nt], false, false);
    }

#pragma unroll
    for (int nt = 0; nt < 4; ++nt) {
      const int col  = n0 + nt * 16 + (lane & 15);
      const float bb = b2[base * kD + col];
#pragma unroll
      for (int mt = 0; mt < 4; ++mt)
#pragma unroll
        for (int v2 = 0; v2 < 8; ++v2) {
          int r = mt * 16 + (lane >> 4) * 8 + v2;
          sC[r * kCSTR + 1024 + col] = f32_to_bf16(acc[mt][nt][v2] + bb);
        }
    }
  }
  __syncthreads();

  // ---- stage 4: out = relu(concat @ Wf + bf) ----
  {
    v8f acc[4][4];
#pragma unroll
    for (int mt = 0; mt < 4; ++mt)
#pragma unroll
      for (int nt = 0; nt < 4; ++nt) acc[mt][nt] = vzero;

    for (int kt = 0; kt < (3 * kD) / 32; ++kt) {
      const int k0 = kt * 32;
      Frag a[4], b[4];
#pragma unroll
      for (int mt = 0; mt < 4; ++mt) {
        const unsigned short* rp = sC + (mt * 16 + mrow) * kCSTR + k0 + khlf * 16;
        a[mt].q[0] = *(const uint4*)(rp);
        a[mt].q[1] = *(const uint4*)(rp + 8);
      }
#pragma unroll
      for (int nt = 0; nt < 4; ++nt) {
        const unsigned short* bp =
            Wfs + ((((wv * 4 + nt) * 48) + kt) << 9) + lane * 16;
        b[nt].q[0] = *(const uint4*)(bp);
        b[nt].q[1] = *(const uint4*)(bp + 8);
        __builtin_prefetch(bp + 512, 0, 0);
      }
#pragma unroll
      for (int mt = 0; mt < 4; ++mt)
#pragma unroll
        for (int nt = 0; nt < 4; ++nt)
          acc[mt][nt] = __builtin_amdgcn_wmma_f32_16x16x32_bf16(
              false, a[mt].v, false, b[nt].v, (short)0, acc[mt][nt], false, false);
    }

#pragma unroll
    for (int nt = 0; nt < 4; ++nt) {
      const int col  = n0 + nt * 16 + (lane & 15);
      const float bb = bfv[col];
#pragma unroll
      for (int mt = 0; mt < 4; ++mt)
#pragma unroll
        for (int v2 = 0; v2 < 8; ++v2) {
          int r = mt * 16 + (lane >> 4) * 8 + v2;
          if (row0 + r < cnt)
            out[sIdx[r] * kD + col] = fmaxf(acc[mt][nt][v2] + bb, 0.f);
        }
    }
  }
}

// ---------------- launch ----------------
extern "C" void kernel_launch(void* const* d_in, const int* in_sizes, int n_in,
                              void* d_out, int out_size, void* d_ws, size_t ws_size,
                              hipStream_t stream) {
  const int*   type_ids     = (const int*)d_in[0];
  const int*   source_ids   = (const int*)d_in[1];
  const float* params       = (const float*)d_in[2];
  const float* type_embed   = (const float*)d_in[3];
  const float* source_embed = (const float*)d_in[4];
  const float* W1           = (const float*)d_in[5];
  const float* b1           = (const float*)d_in[6];
  const float* W2           = (const float*)d_in[7];
  const float* b2           = (const float*)d_in[8];
  const float* Wf           = (const float*)d_in[9];
  const float* bfv          = (const float*)d_in[10];
  float* out = (float*)d_out;

  char* ws = (char*)d_ws;
  int* counts  = (int*)ws;                                   // 16 ints (256 B slot)
  int* buckets = (int*)(ws + 256);                           // 9 * 65536 ints
  unsigned short* W2s = (unsigned short*)(ws + 256 + (size_t)kNBASE * kE * 4);
  unsigned short* Wfs = W2s + (size_t)kNBASE * kD * kD;      // then 1536*512 halves

  zero_counts_kernel<<<1, 32, 0, stream>>>(counts);
  bin_edges_kernel<<<kE / 256, 256, 0, stream>>>(type_ids, counts, buckets);
  swz_w2_kernel<<<(kNBASE * kD * kD + 255) / 256, 256, 0, stream>>>(W2, W2s);
  swz_wf_kernel<<<(3 * kD * kD + 255) / 256, 256, 0, stream>>>(Wf, Wfs);

  const int smem = (kROWS * kCSTR + kROWS * kHSTR) * 2       // concat + h (bf16)
                 + kROWS * kPMAX * 4                         // x
                 + kROWS * 3 * 4                             // idx/tid/sid
                 + 128;
  (void)hipFuncSetAttribute((const void*)edge_mlp_kernel,
                            hipFuncAttributeMaxDynamicSharedMemorySize, smem);
  edge_mlp_kernel<<<kNBASE * kTILES_PER_BASE, 256, smem, stream>>>(
      type_ids, source_ids, params, type_embed, source_embed,
      W1, b1, b2, bfv, counts, buckets, W2s, Wfs, out);
}
